// GNNSwarmPolicy_16458314678415
// MI455X (gfx1250) — matrix operand order repaired and checked by reading, not compiled, not run on hardware
//
#include <hip/hip_runtime.h>
#include <hip/hip_bf16.h>
#include <math.h>

// ---------------------------------------------------------------------------
// Problem constants (from reference)
// ---------------------------------------------------------------------------
#define BB    8
#define NN    4096
#define BN    32768          // BB*NN
#define KNB   10             // K nearest neighbors
#define HID   128
#define HEADS 4
#define DH    32
#define PHYS  9
#define EDGES (BN * (KNB + 1))   // + self loops = 360448
#define EPSLN 1e-5f
#define SLOPE 0.2f

// padded LDS row stride (halves): 128 + 8 -> 272B = 17*16B (16B aligned rows,
// 4-bank rotation per row => conflict-free 16B fragment loads)
#define LPAD  136
#define AOFF  (64 * LPAD)        // sA elements before sW region

typedef __attribute__((ext_vector_type(16))) _Float16 v16h;
typedef __attribute__((ext_vector_type(8)))  _Float16 v8h;
typedef __attribute__((ext_vector_type(8)))  float    v8f;
typedef __attribute__((ext_vector_type(4)))  unsigned u32x4;
typedef __attribute__((ext_vector_type(8)))  int      i32x8;
typedef __attribute__((ext_vector_type(4)))  int      i32x4;

// ---------------- ordered-uint float encoding for atomicMax -----------------
__device__ __forceinline__ unsigned f2ord(float f) {
    unsigned u = __float_as_uint(f);
    return (u & 0x80000000u) ? ~u : (u | 0x80000000u);
}
__device__ __forceinline__ float ord2f(unsigned u) {
    unsigned v = (u & 0x80000000u) ? (u & 0x7FFFFFFFu) : ~u;
    return __uint_as_float(v);
}

// ---------------------------------------------------------------------------
// TDM: DMA a rows x 128 f16 tile (contiguous rows, stride 128 elems) from
// global into LDS at byte offset lds_off, padding +16B per 256B row so the
// LDS row stride becomes 272B (= LPAD halves).
//   D# group0: count=1 | lds_addr | global_addr | type=2
//   D# group1: data_size=2B, pad_enable, pad_interval=64 dwords (code 5),
//              pad_amount=4 dwords (code 3), dims/strides as below
// (6-arg builtin form on this toolchain: g0, g1, g2, g3, g4, cpol)
// ---------------------------------------------------------------------------
__device__ __forceinline__ void tdm_load_tile_f16(unsigned lds_off,
                                                  const _Float16* g,
                                                  unsigned rows) {
    unsigned long long ga = (unsigned long long)(const void*)g;
    u32x4 g0 = {};
    g0[0] = 1u;                                        // count=1, user mode
    g0[1] = lds_off;                                   // lds_addr (bytes)
    g0[2] = (unsigned)ga;                              // global_addr[31:0]
    g0[3] = (unsigned)((ga >> 32) & 0x01FFFFFFull) | (2u << 30);  // ga[56:32] | type=2
    i32x8 g1 = {};
    g1[0] = (int)((1u << 16)      // data_size = 1 -> 2 bytes
                | (1u << 20)      // pad_enable
                | (5u << 22)      // pad_interval code 5 = 64 dwords (256B)
                | (3u << 25));    // pad_amount  code 3 = 4 dwords (16B)
    g1[1] = (int)(128u << 16);    // tensor_dim0 = 128 (low 16)
    g1[2] = (int)(rows << 16);    // tensor_dim0 hi = 0 | tensor_dim1 lo = rows
    g1[3] = (int)(128u << 16);    // tensor_dim1 hi = 0 | tile_dim0 = 128
    g1[4] = (int)rows;            // tile_dim1 = rows, tile_dim2 = 0
    g1[5] = 128;                  // tensor_dim0_stride[31:0] = 128 elems
    i32x4 z4 = {};
    i32x8 z8 = {};
    __builtin_amdgcn_tensor_load_to_lds(g0, g1, z4, z4, z8, 0);
}

// ---------------------------------------------------------------------------
// 1) Encoder: x = LN(relu(phys @ encW + b)) + silu(lat*temW + temb)  -> f16 x
// ---------------------------------------------------------------------------
__global__ void enc_kernel(const float* __restrict__ obs,
                           const float* __restrict__ encW,
                           const float* __restrict__ encB,
                           const float* __restrict__ lnG,
                           const float* __restrict__ lnB,
                           const float* __restrict__ temW,
                           const float* __restrict__ temB,
                           _Float16* __restrict__ x) {
    __shared__ float sobs[16];
    __shared__ float red[HID];
    const int g = blockIdx.x;
    const int t = threadIdx.x;
    if (t < 10) sobs[t] = obs[(size_t)g * 10 + t];
    __syncthreads();

    float acc = encB[t];
#pragma unroll
    for (int j = 0; j < PHYS; ++j) acc = fmaf(sobs[j], encW[j * HID + t], acc);
    float y = fmaxf(acc, 0.f);

    red[t] = y; __syncthreads();
    for (int s = 64; s > 0; s >>= 1) { if (t < s) red[t] += red[t + s]; __syncthreads(); }
    const float mean = red[0] * (1.f / HID);
    __syncthreads();
    const float d = y - mean;
    red[t] = d * d; __syncthreads();
    for (int s = 64; s > 0; s >>= 1) { if (t < s) red[t] += red[t + s]; __syncthreads(); }
    const float var = red[0] * (1.f / HID);
    float xn = d * rsqrtf(var + EPSLN) * lnG[t] + lnB[t];

    const float l = fmaf(sobs[9], temW[t], temB[t]);
    const float silu = l / (1.f + __expf(-l));
    x[(size_t)g * HID + t] = (_Float16)(xn + silu);
}

// ---------------------------------------------------------------------------
// 2) KNN top-10 (squared distance), LDS-tiled positions
// ---------------------------------------------------------------------------
__global__ void knn_kernel(const float* __restrict__ obs,
                           int* __restrict__ knn_idx) {
    __shared__ float sx[256], sy[256], sz[256];
    const int g = blockIdx.x * 256 + threadIdx.x;
    const int b = g >> 12;
    const int i = g & (NN - 1);
    const float* base = obs + (size_t)b * NN * 10;
    const float px = base[i * 10 + 0];
    const float py = base[i * 10 + 1];
    const float pz = base[i * 10 + 2];

    float bd[KNB]; int bi[KNB];
#pragma unroll
    for (int q = 0; q < KNB; ++q) { bd[q] = 3.0e38f; bi[q] = i; }

    for (int t0 = 0; t0 < NN; t0 += 256) {
        const int jl = t0 + threadIdx.x;
        sx[threadIdx.x] = base[jl * 10 + 0];
        sy[threadIdx.x] = base[jl * 10 + 1];
        sz[threadIdx.x] = base[jl * 10 + 2];
        __syncthreads();
        for (int jj = 0; jj < 256; ++jj) {
            const int j2 = t0 + jj;
            const float dx = px - sx[jj];
            const float dy = py - sy[jj];
            const float dz = pz - sz[jj];
            float d2 = fmaf(dx, dx, fmaf(dy, dy, dz * dz));
            if (j2 == i) d2 = 3.0e38f;
            if (d2 < bd[KNB - 1]) {
                bd[KNB - 1] = d2; bi[KNB - 1] = j2;
#pragma unroll
                for (int q = KNB - 1; q > 0; --q) {
                    if (bd[q] < bd[q - 1]) {
                        float td = bd[q]; bd[q] = bd[q - 1]; bd[q - 1] = td;
                        int   ti = bi[q]; bi[q] = bi[q - 1]; bi[q - 1] = ti;
                    }
                }
            }
        }
        __syncthreads();
    }
#pragma unroll
    for (int q = 0; q < KNB; ++q) knn_idx[(size_t)g * KNB + q] = b * NN + bi[q];
}

// ---------------------------------------------------------------------------
// 3) weight prep: Wt[c][r] = (f16) W[r][c]   (128x128)
// ---------------------------------------------------------------------------
__global__ void wprep_kernel(const float* __restrict__ W,
                             _Float16* __restrict__ Wt) {
    const int idx = blockIdx.x * 256 + threadIdx.x;   // < 16384
    const int r = idx >> 7, c = idx & 127;
    Wt[c * HID + r] = (_Float16)W[idx];
}

// ---------------------------------------------------------------------------
// 4) WMMA GEMM: C[M x 128] = A_f16[M x 128] @ W[128 x 128] (+bias, +relu)
//    A and Wt (pre-transposed) staged to LDS via TDM; 8 waves; 64x128 tile
// ---------------------------------------------------------------------------
__global__ void gemm128_kernel(const _Float16* __restrict__ A,
                               const _Float16* __restrict__ Wt,
                               const float* __restrict__ bias,
                               float* __restrict__ C,
                               int relu) {
    // [0, AOFF)            : sA   64 rows  x LPAD halves
    // [AOFF, AOFF+128*LPAD): sWt 128 cols x LPAD halves (Wt[col][k])
    __shared__ __align__(16) _Float16 smem[AOFF + 128 * LPAD];

    const int mbase = blockIdx.x * 64;

    if (threadIdx.x == 0) {
        tdm_load_tile_f16(0u, A + (size_t)mbase * HID, 64u);
        tdm_load_tile_f16((unsigned)(AOFF * sizeof(_Float16)), Wt, 128u);
        __builtin_amdgcn_s_wait_tensorcnt(0);
    }
    __syncthreads();
    // The TDM engine wrote smem behind the compiler's back; escape the pointer
    // with a memory clobber so the LDS fragment loads below are not folded.
    {
        _Float16* p = smem;
        asm volatile("" : "+v"(p) : : "memory");
    }

    const int lane = threadIdx.x & 31;
    const int wave = threadIdx.x >> 5;
    const int m16  = (wave & 3) * 16;     // M sub-tile base within 64
    const int nh   = (wave >> 2) * 64;    // N half base
    const int hl   = lane >> 4;           // lane half (0/1)
    const int l16  = lane & 15;

    v8f acc[4] = {};

    const int arow = (m16 + l16) * LPAD;

#pragma unroll
    for (int ks = 0; ks < 4; ++ks) {
        const int kb = ks * 32;
        // A fragment (16-bit A 16x32): lanes 0-15 K{0-7,16-23}, 16-31 K{8-15,24-31}
        const int ak = kb + hl * 8;
        v8h alo = *(const v8h*)&smem[arow + ak];
        v8h ahi = *(const v8h*)&smem[arow + ak + 16];
        v16h afrag = __builtin_shufflevector(alo, ahi,
            0, 1, 2, 3, 4, 5, 6, 7, 8, 9, 10, 11, 12, 13, 14, 15);
#pragma unroll
        for (int nt = 0; nt < 4; ++nt) {
            const int col = nh + nt * 16 + l16;
            const int bk  = kb + hl * 8;      // B: lanes 0-15 K{0-7,16-23}, 16-31 K{8-15,24-31}
            const int bbase = AOFF + col * LPAD + bk;
            v8h blo = *(const v8h*)&smem[bbase];
            v8h bhi = *(const v8h*)&smem[bbase + 16];
            v16h bfrag = __builtin_shufflevector(blo, bhi,
                0, 1, 2, 3, 4, 5, 6, 7, 8, 9, 10, 11, 12, 13, 14, 15);
            acc[nt] = __builtin_amdgcn_wmma_f32_16x16x32_f16(
                false, afrag, false, bfrag, (short)0, acc[nt], false, false);
        }
    }

    // C/D layout: VGPR v -> row = v + 8*half, col = lane%16
#pragma unroll
    for (int nt = 0; nt < 4; ++nt) {
        const int col = nh + nt * 16 + l16;
        const float bv = bias ? bias[col] : 0.f;
#pragma unroll
        for (int v = 0; v < 8; ++v) {
            const int row = mbase + m16 + hl * 8 + v;
            float val = acc[nt][v] + bv;
            if (relu) val = fmaxf(val, 0.f);
            C[(size_t)row * HID + col] = val;
        }
    }
}

// ---------------------------------------------------------------------------
// 5) attention prep: as_/ad_[node,head] = sum_d h*a
// ---------------------------------------------------------------------------
__global__ void attn_prep_kernel(const float* __restrict__ h,
                                 const float* __restrict__ a_src,
                                 const float* __restrict__ a_dst,
                                 float* __restrict__ as_,
                                 float* __restrict__ ad_) {
    const int t = blockIdx.x * 256 + threadIdx.x;
    const int g = t >> 2, hh = t & 3;
    const float* hr = h + (size_t)g * HID + hh * DH;
    float sa = 0.f, sd = 0.f;
#pragma unroll 8
    for (int d = 0; d < DH; ++d) {
        const float v = hr[d];
        sa = fmaf(v, a_src[hh * DH + d], sa);
        sd = fmaf(v, a_dst[hh * DH + d], sd);
    }
    as_[t] = sa; ad_[t] = sd;
}

// ---------------------------------------------------------------------------
// 6) per-layer init of msg / ssum / mmax
// ---------------------------------------------------------------------------
__global__ void seg_init_kernel(float* __restrict__ msg,
                                float* __restrict__ ssum,
                                unsigned* __restrict__ mmax) {
    const int t = blockIdx.x * 256 + threadIdx.x;
    if (t < BN * HID) msg[t] = 0.f;
    if (t < BN * HEADS) { ssum[t] = 0.f; mmax[t] = f2ord(-3.0e38f); }
}

// ---------------------------------------------------------------------------
// 7) edge pass 1: e = leaky_relu(as_[src]+ad_[dst]); segment max into mmax
// ---------------------------------------------------------------------------
__global__ void edge_max_kernel(const int* __restrict__ knn_idx,
                                const float* __restrict__ as_,
                                const float* __restrict__ ad_,
                                float* __restrict__ eattn,
                                unsigned* __restrict__ mmax) {
    const int t = blockIdx.x * 256 + threadIdx.x;
    const int e = t >> 2, hh = t & 3;
    const int g = e / (KNB + 1);
    const int k = e - g * (KNB + 1);
    const int src = g;
    const int dst = (k < KNB) ? knn_idx[(size_t)g * KNB + k] : g;
    float v = as_[src * HEADS + hh] + ad_[dst * HEADS + hh];
    v = (v > 0.f) ? v : SLOPE * v;
    eattn[t] = v;
    atomicMax(&mmax[dst * HEADS + hh], f2ord(v));
}

// ---------------------------------------------------------------------------
// 8) edge pass 2: ex = exp(e-m); ssum += ex; msg[dst] += ex * h[src]
// ---------------------------------------------------------------------------
__global__ void edge_acc_kernel(const int* __restrict__ knn_idx,
                                const float* __restrict__ eattn,
                                const unsigned* __restrict__ mmax,
                                const float* __restrict__ h,
                                float* __restrict__ ssum,
                                float* __restrict__ msg) {
    const int t = blockIdx.x * 256 + threadIdx.x;
    const int e = t >> 2, hh = t & 3;
    const int g = e / (KNB + 1);
    const int k = e - g * (KNB + 1);
    const int src = g;
    const int dst = (k < KNB) ? knn_idx[(size_t)g * KNB + k] : g;
    const float m = ord2f(mmax[dst * HEADS + hh]);
    const float ex = __expf(eattn[t] - m);
    atomicAdd(&ssum[dst * HEADS + hh], ex);
    const float* hr = h + (size_t)src * HID + hh * DH;
    float* mo = msg + (size_t)dst * HID + hh * DH;
#pragma unroll 8
    for (int d = 0; d < DH; ++d) atomicAdd(&mo[d], ex * hr[d]);
}

// ---------------------------------------------------------------------------
// 9) GAT post: x = relu(LN(msg/ssum + bias))  -> f16 x
// ---------------------------------------------------------------------------
__global__ void gat_post_kernel(const float* __restrict__ msg,
                                const float* __restrict__ ssum,
                                const float* __restrict__ gb,
                                const float* __restrict__ lnG,
                                const float* __restrict__ lnB,
                                _Float16* __restrict__ x) {
    __shared__ float red[HID];
    const int g = blockIdx.x;
    const int t = threadIdx.x;
    const int hh = t >> 5;
    const float val = msg[(size_t)g * HID + t] / ssum[g * HEADS + hh] + gb[t];

    red[t] = val; __syncthreads();
    for (int s = 64; s > 0; s >>= 1) { if (t < s) red[t] += red[t + s]; __syncthreads(); }
    const float mean = red[0] * (1.f / HID);
    __syncthreads();
    const float d = val - mean;
    red[t] = d * d; __syncthreads();
    for (int s = 64; s > 0; s >>= 1) { if (t < s) red[t] += red[t + s]; __syncthreads(); }
    const float var = red[0] * (1.f / HID);
    const float y = d * rsqrtf(var + EPSLN) * lnG[t] + lnB[t];
    x[(size_t)g * HID + t] = (_Float16)fmaxf(y, 0.f);
}

// ---------------------------------------------------------------------------
// 10) heads
// ---------------------------------------------------------------------------
__global__ void actor_out_kernel(const float* __restrict__ hid,
                                 const float* __restrict__ W2,
                                 const float* __restrict__ b2,
                                 float* __restrict__ out) {
    const int g = blockIdx.x * 256 + threadIdx.x;
    const float* hr = hid + (size_t)g * HID;
    float a0 = b2[0], a1 = b2[1], a2 = b2[2];
#pragma unroll 8
    for (int d = 0; d < HID; ++d) {
        const float v = hr[d];
        a0 = fmaf(v, W2[d * 3 + 0], a0);
        a1 = fmaf(v, W2[d * 3 + 1], a1);
        a2 = fmaf(v, W2[d * 3 + 2], a2);
    }
    out[(size_t)g * 3 + 0] = tanhf(a0);
    out[(size_t)g * 3 + 1] = tanhf(a1);
    out[(size_t)g * 3 + 2] = tanhf(a2);
}

__global__ void critic_out_kernel(const float* __restrict__ hid,
                                  const float* __restrict__ W2,
                                  const float* __restrict__ b2,
                                  float* __restrict__ out) {
    const int g = blockIdx.x * 256 + threadIdx.x;
    const float* hr = hid + (size_t)g * HID;
    float a = b2[0];
#pragma unroll 8
    for (int d = 0; d < HID; ++d) a = fmaf(hr[d], W2[d], a);
    out[g] = a;
}

// ---------------------------------------------------------------------------
// launch — inputs in setup_inputs() dict order:
//  0 obs, 1 enc_W, 2 enc_b, 3 enc_ln_g, 4 enc_ln_b, 5 tem_W, 6 tem_b,
//  7 actor_W1, 8 actor_b1, 9 actor_W2, 10 actor_b2,
//  11 critic_W1, 12 critic_b1, 13 critic_W2, 14 critic_b2,
//  15..20 gat0_{W,asrc,adst,b,ln_g,ln_b}, 21..26 gat1_{...}
// ---------------------------------------------------------------------------
extern "C" void kernel_launch(void* const* d_in, const int* in_sizes, int n_in,
                              void* d_out, int out_size, void* d_ws, size_t ws_size,
                              hipStream_t stream) {
    (void)in_sizes; (void)n_in; (void)out_size; (void)ws_size;
    const float* obs   = (const float*)d_in[0];
    const float* encW  = (const float*)d_in[1];
    const float* encB  = (const float*)d_in[2];
    const float* encG  = (const float*)d_in[3];
    const float* encLB = (const float*)d_in[4];
    const float* temW  = (const float*)d_in[5];
    const float* temB  = (const float*)d_in[6];
    const float* aW1   = (const float*)d_in[7];
    const float* ab1   = (const float*)d_in[8];
    const float* aW2   = (const float*)d_in[9];
    const float* ab2   = (const float*)d_in[10];
    const float* cW1   = (const float*)d_in[11];
    const float* cb1   = (const float*)d_in[12];
    const float* cW2   = (const float*)d_in[13];
    const float* cb2   = (const float*)d_in[14];
    const float* gW[2]    = {(const float*)d_in[15], (const float*)d_in[21]};
    const float* gasrc[2] = {(const float*)d_in[16], (const float*)d_in[22]};
    const float* gadst[2] = {(const float*)d_in[17], (const float*)d_in[23]};
    const float* gb[2]    = {(const float*)d_in[18], (const float*)d_in[24]};
    const float* gg[2]    = {(const float*)d_in[19], (const float*)d_in[25]};
    const float* glb[2]   = {(const float*)d_in[20], (const float*)d_in[26]};

    const size_t MB = 1u << 20;
    char* ws = (char*)d_ws;
    _Float16* x     = (_Float16*)(ws + 0 * MB);   // BN*128 f16 = 8 MB
    float*    h     = (float*)(ws + 16 * MB);     // BN*128 f32 = 16 MB
    float*    msg   = (float*)(ws + 32 * MB);     // BN*128 f32 = 16 MB
    float*    as_   = (float*)(ws + 48 * MB);     // BN*4
    float*    ad_   = (float*)(ws + 49 * MB);     // BN*4
    float*    ssum  = (float*)(ws + 50 * MB);     // BN*4
    unsigned* mmax  = (unsigned*)(ws + 51 * MB);  // BN*4
    float*    eattn = (float*)(ws + 52 * MB);     // EDGES*4 f32 = 5.5 MB
    int*      knn   = (int*)(ws + 58 * MB);       // BN*10 int
    _Float16* WtG0  = (_Float16*)(ws + 60 * MB);                // 32 KB each
    _Float16* WtG1  = (_Float16*)(ws + 60 * MB + 64 * 1024);
    _Float16* WtA   = (_Float16*)(ws + 60 * MB + 128 * 1024);
    _Float16* WtC   = (_Float16*)(ws + 60 * MB + 192 * 1024);
    _Float16* WtL[2] = {WtG0, WtG1};

    float* act_out = (float*)d_out;               // (B,N,3)
    float* val_out = (float*)d_out + (size_t)BN * 3;

    // weight prep (f32 -> f16 transposed)
    wprep_kernel<<<64, 256, 0, stream>>>(gW[0], WtG0);
    wprep_kernel<<<64, 256, 0, stream>>>(gW[1], WtG1);
    wprep_kernel<<<64, 256, 0, stream>>>(aW1, WtA);
    wprep_kernel<<<64, 256, 0, stream>>>(cW1, WtC);

    // 1) encoder
    enc_kernel<<<BN, HID, 0, stream>>>(obs, encW, encB, encG, encLB, temW, temB, x);
    // 2) knn
    knn_kernel<<<BN / 256, 256, 0, stream>>>(obs, knn);

    // 3) GAT layers
    for (int l = 0; l < 2; ++l) {
        gemm128_kernel<<<BN / 64, 256, 0, stream>>>(x, WtL[l], nullptr, h, 0);
        attn_prep_kernel<<<BN * HEADS / 256, 256, 0, stream>>>(h, gasrc[l], gadst[l], as_, ad_);
        seg_init_kernel<<<BN * HID / 256, 256, 0, stream>>>(msg, ssum, mmax);
        edge_max_kernel<<<EDGES * HEADS / 256, 256, 0, stream>>>(knn, as_, ad_, eattn, mmax);
        edge_acc_kernel<<<EDGES * HEADS / 256, 256, 0, stream>>>(knn, eattn, mmax, h, ssum, msg);
        gat_post_kernel<<<BN, HID, 0, stream>>>(msg, ssum, gb[l], gg[l], glb[l], x);
    }

    // 4) actor head
    gemm128_kernel<<<BN / 64, 256, 0, stream>>>(x, WtA, ab1, h, 1);
    actor_out_kernel<<<BN / 256, 256, 0, stream>>>(h, aW2, ab2, act_out);
    // 5) critic head
    gemm128_kernel<<<BN / 64, 256, 0, stream>>>(x, WtC, cb1, h, 1);
    critic_out_kernel<<<BN / 256, 256, 0, stream>>>(h, cW2, cb2, val_out);
}
